// SplitTransformer_78271484002638
// MI455X (gfx1250) — matrix operand order repaired and checked by reading, not compiled, not run on hardware
//
#include <hip/hip_runtime.h>
#include <hip/hip_bf16.h>

typedef _Float16 v8h  __attribute__((ext_vector_type(8)));
typedef _Float16 v16h __attribute__((ext_vector_type(16)));
typedef float    v8f  __attribute__((ext_vector_type(8)));
typedef unsigned int u32x4 __attribute__((ext_vector_type(4)));
typedef int          i32x4 __attribute__((ext_vector_type(4)));
typedef int          i32x8 __attribute__((ext_vector_type(8)));

#define B_     8
#define N_     1024
#define D_     768
#define H_     12
#define DH_    64
#define L_     4
#define MLP_   3072
#define INNER_ 768

// ---------------------------------------------------------------------------
// Elementwise kernels
// ---------------------------------------------------------------------------
__global__ void cvt_f32_f16_kernel(const float* __restrict__ in,
                                   _Float16* __restrict__ out, long n) {
  long i = (long)blockIdx.x * blockDim.x + threadIdx.x;
  long stride = (long)gridDim.x * blockDim.x;
  for (; i < n; i += stride) out[i] = (_Float16)in[i];
}

// One block (256 threads) per row of length D; biased variance like jnp.var.
__global__ void ln_kernel(const float* __restrict__ x, const float* __restrict__ w,
                          const float* __restrict__ b, _Float16* __restrict__ out,
                          int D) {
  __shared__ float s1[256], s2[256];
  const long row = blockIdx.x;
  const float* xr = x + row * (long)D;
  float a = 0.f, q = 0.f;
  for (int i = threadIdx.x; i < D; i += 256) {
    float v = xr[i];
    a += v;
    q += v * v;
  }
  s1[threadIdx.x] = a;
  s2[threadIdx.x] = q;
  __syncthreads();
  for (int o = 128; o > 0; o >>= 1) {
    if ((int)threadIdx.x < o) {
      s1[threadIdx.x] += s1[threadIdx.x + o];
      s2[threadIdx.x] += s2[threadIdx.x + o];
    }
    __syncthreads();
  }
  float mean = s1[0] / (float)D;
  float var  = s2[0] / (float)D - mean * mean;
  float inv  = rsqrtf(var + 1e-5f);
  _Float16* orow = out + row * (long)D;
  for (int i = threadIdx.x; i < D; i += 256)
    orow[i] = (_Float16)((xr[i] - mean) * inv * w[i] + b[i]);
}

// Row softmax over fixed length N_=1024, 4 elems/thread, f32 in -> f16 out.
__global__ void softmax_kernel(const float* __restrict__ sc,
                               _Float16* __restrict__ out) {
  __shared__ float sm[256];
  const long row = blockIdx.x;
  const float* r = sc + row * (long)N_;
  float v[4];
  float m = -3.4e38f;
#pragma unroll
  for (int j = 0; j < 4; ++j) {
    v[j] = r[threadIdx.x + j * 256];
    m = fmaxf(m, v[j]);
  }
  sm[threadIdx.x] = m;
  __syncthreads();
  for (int o = 128; o > 0; o >>= 1) {
    if ((int)threadIdx.x < o)
      sm[threadIdx.x] = fmaxf(sm[threadIdx.x], sm[threadIdx.x + o]);
    __syncthreads();
  }
  float rowMax = sm[0];
  __syncthreads();
  float s = 0.f;
#pragma unroll
  for (int j = 0; j < 4; ++j) {
    v[j] = __expf(v[j] - rowMax);
    s += v[j];
  }
  sm[threadIdx.x] = s;
  __syncthreads();
  for (int o = 128; o > 0; o >>= 1) {
    if ((int)threadIdx.x < o) sm[threadIdx.x] += sm[threadIdx.x + o];
    __syncthreads();
  }
  float inv = 1.0f / sm[0];
  _Float16* orow = out + row * (long)N_;
#pragma unroll
  for (int j = 0; j < 4; ++j)
    orow[threadIdx.x + j * 256] = (_Float16)(v[j] * inv);
}

// ---------------------------------------------------------------------------
// Tensor Data Mover: 2D tile (rows x 32 halves, row stride K halves) -> LDS.
// Descriptor per CDNA5 ISA: group0 = {count|flags, lds_addr, gaddr_lo,
// gaddr_hi|type=2}, group1 = {data_size, tensor_dim0/1, tile_dim0/1,
// tensor_dim0_stride}, groups 2/3 zero (2D tensor).
// ---------------------------------------------------------------------------
__device__ __forceinline__ unsigned lds_offset_of(const void* p) {
  // flat address of an LDS object: addr[31:0] is the LDS byte offset
  return (unsigned)(unsigned long long)p;
}

__device__ __forceinline__ void tdm_load_2d(unsigned ldsOff, const _Float16* tile,
                                            int K, int rows) {
  unsigned long long ga = (unsigned long long)tile;
  u32x4 g0;
  g0[0] = 1u;                                                  // count=1, user D#
  g0[1] = ldsOff;                                              // lds_addr
  g0[2] = (unsigned)ga;                                        // global_addr[31:0]
  g0[3] = (unsigned)((ga >> 32) & 0x01FFFFFFu) | 0x80000000u;  // [56:32] | type=2
  i32x8 g1;
  g1[0] = 1 << 16;                                   // data_size = 1 (2 bytes)
  g1[1] = (int)((unsigned)K << 16);                  // tensor_dim0[15:0]
  g1[2] = (int)(((unsigned)K >> 16) |                // tensor_dim0[31:16]
                ((unsigned)rows << 16));             // tensor_dim1[15:0]
  g1[3] = (int)(32u << 16);                          // tensor_dim1[31:16]=0 | tile_dim0=32
  g1[4] = rows;                                      // tile_dim1 | tile_dim2=0
  g1[5] = K;                                         // tensor_dim0_stride[31:0]
  g1[6] = 0;                                         // stride[47:32] | dim1_stride lo
  g1[7] = 0;
  i32x4 z4 = {0, 0, 0, 0};
#if __has_include(<hip/amd_detail/amd_gfx1250_TDM.h>)
  i32x8 z8 = {0, 0, 0, 0, 0, 0, 0, 0};
  __builtin_amdgcn_tensor_load_to_lds(g0, g1, z4, z4, z8, 0);
#else
  __builtin_amdgcn_tensor_load_to_lds(g0, g1, z4, z4, 0);
#endif
}

// ---------------------------------------------------------------------------
// WMMA GEMM:  C[m,n] = sum_k A[m,k] * W[n,k]   (A: MxK, W: NxK, both K-major)
// Block = 256 threads = 8 waves (wave32), wave tile 64x32 (4x2 16x16 frags),
// block tile 128x128.  K-slabs staged to LDS by the TDM (double buffered;
// all K used here are multiples of 64 so the 2-step unroll has no tail).
// Batched via blockIdx.z with (z/zdiv, z%zdiv) strides.
// Epilogues: 0=f32*scale  1=f16  2=f16 transposed (C[col*ldc+row])
//            3=f32+bias   4=f16 gelu(x+bias)   5=f32+bias+resid
// ---------------------------------------------------------------------------
__device__ __forceinline__ void wmma_step(const _Float16* __restrict__ sa,
                                          const _Float16* __restrict__ sw,
                                          const int* aRowL, const int* wRowL,
                                          int half, v8f acc[4][2]) {
  v16h bfrag[2];
#pragma unroll
  for (int nf = 0; nf < 2; ++nf)
    bfrag[nf] = *(const v16h*)(sw + wRowL[nf] * 32 + half * 16);
#pragma unroll
  for (int mf = 0; mf < 4; ++mf) {
    v8h lo = *(const v8h*)(sa + aRowL[mf] * 32 + half * 16);
    v8h hi = *(const v8h*)(sa + aRowL[mf] * 32 + half * 16 + 8);
    v16h af;
#pragma unroll
    for (int i = 0; i < 8; ++i) {
      af[i]     = lo[i];
      af[i + 8] = hi[i];
    }
#pragma unroll
    for (int nf = 0; nf < 2; ++nf)
      acc[mf][nf] = __builtin_amdgcn_wmma_f32_16x16x32_f16(
          false, af, false, bfrag[nf], (short)0, acc[mf][nf], false, false);
  }
}

template <int EPI>
__global__ void __launch_bounds__(256)
gemm_wmma_kernel(const _Float16* __restrict__ A, const _Float16* __restrict__ W,
                 const float* __restrict__ bias, const float* __restrict__ resid,
                 void* __restrict__ C, int M, int N, int K, int ldc, float scale,
                 int zdiv, long aOuter, long aInner, long wOuter, long wInner,
                 long cOuter, long cInner) {
  __shared__ __align__(64) _Float16 smA[2][128 * 32];
  __shared__ __align__(64) _Float16 smW[2][128 * 32];

  const int lane = threadIdx.x & 31;
  const int wave = threadIdx.x >> 5;
  const int l16  = lane & 15;
  const int half = lane >> 4;
  const int blockM = blockIdx.y * 128;
  const int blockN = blockIdx.x * 128;
  const int z  = blockIdx.z;
  const int zo = z / zdiv;
  const int zi = z % zdiv;
  const _Float16* Ab = A + (long)zo * aOuter + (long)zi * aInner;
  const _Float16* Wb = W + (long)zo * wOuter + (long)zi * wInner;
  const long cBase = (long)zo * cOuter + (long)zi * cInner;

  const int rowsA = (M - blockM < 128) ? (M - blockM) : 128;
  const int rowsW = (N - blockN < 128) ? (N - blockN) : 128;
  const _Float16* Atile = Ab + (long)blockM * K;
  const _Float16* Wtile = Wb + (long)blockN * K;

  // block-local fragment rows (clamped against the loaded slab)
  int aRowL[4], wRowL[2];
#pragma unroll
  for (int mf = 0; mf < 4; ++mf) {
    int r = (wave >> 2) * 64 + mf * 16 + l16;
    if (r >= rowsA) r = rowsA - 1;
    aRowL[mf] = r;
  }
#pragma unroll
  for (int nf = 0; nf < 2; ++nf) {
    int c = (wave & 3) * 32 + nf * 16 + l16;
    if (c >= rowsW) c = rowsW - 1;
    wRowL[nf] = c;
  }

  v8f acc[4][2] = {};
  const int nsteps = K >> 5;  // always even for K in {768,1024,3072}

  if (threadIdx.x < 32) {  // wave 0 drives the TDM (EXEC-independent DMA)
    tdm_load_2d(lds_offset_of(&smA[0][0]), Atile, K, rowsA);
    tdm_load_2d(lds_offset_of(&smW[0][0]), Wtile, K, rowsW);
  }

  for (int s = 0; s < nsteps; s += 2) {
    __builtin_amdgcn_s_wait_tensorcnt(0);  // buffer 0 slabs resident
    __syncthreads();
    if (threadIdx.x < 32 && s + 1 < nsteps) {
      tdm_load_2d(lds_offset_of(&smA[1][0]), Atile + (s + 1) * 32, K, rowsA);
      tdm_load_2d(lds_offset_of(&smW[1][0]), Wtile + (s + 1) * 32, K, rowsW);
    }
    wmma_step(smA[0], smW[0], aRowL, wRowL, half, acc);

    __builtin_amdgcn_s_wait_tensorcnt(0);  // buffer 1 slabs resident
    __syncthreads();
    if (threadIdx.x < 32 && s + 2 < nsteps) {
      tdm_load_2d(lds_offset_of(&smA[0][0]), Atile + (s + 2) * 32, K, rowsA);
      tdm_load_2d(lds_offset_of(&smW[0][0]), Wtile + (s + 2) * 32, K, rowsW);
    }
    wmma_step(smA[1], smW[1], aRowL, wRowL, half, acc);
  }

#pragma unroll
  for (int mf = 0; mf < 4; ++mf) {
#pragma unroll
    for (int nf = 0; nf < 2; ++nf) {
      const int col = blockN + (wave & 3) * 32 + nf * 16 + l16;
      if (col >= N) continue;
#pragma unroll
      for (int r = 0; r < 8; ++r) {
        const int row = blockM + (wave >> 2) * 64 + mf * 16 + half * 8 + r;
        if (row >= M) continue;
        const float v = acc[mf][nf][r];
        if constexpr (EPI == 0) {
          ((float*)C)[cBase + (long)row * ldc + col] = v * scale;
        } else if constexpr (EPI == 1) {
          ((_Float16*)C)[cBase + (long)row * ldc + col] = (_Float16)v;
        } else if constexpr (EPI == 2) {
          ((_Float16*)C)[cBase + (long)col * ldc + row] = (_Float16)v;
        } else if constexpr (EPI == 3) {
          ((float*)C)[cBase + (long)row * ldc + col] = v + bias[col];
        } else if constexpr (EPI == 4) {
          float t = v + bias[col];
          float g = 0.5f * t * (1.0f + erff(t * 0.70710678118654752f));
          ((_Float16*)C)[cBase + (long)row * ldc + col] = (_Float16)g;
        } else {
          ((float*)C)[cBase + (long)row * ldc + col] =
              v + bias[col] + resid[(long)row * ldc + col];
        }
      }
    }
  }
}

// ---------------------------------------------------------------------------
// Host driver
// ---------------------------------------------------------------------------
extern "C" void kernel_launch(void* const* d_in, const int* in_sizes, int n_in,
                              void* d_out, int out_size, void* d_ws, size_t ws_size,
                              hipStream_t stream) {
  (void)in_sizes; (void)n_in; (void)out_size; (void)ws_size;
  const float* x     = (const float*)d_in[0];
  const float* pos   = (const float*)d_in[1];
  const float* basep = (const float*)d_in[2];
  const float* ln1w  = (const float*)d_in[3];
  const float* ln1b  = (const float*)d_in[4];
  const float* Wqkv  = (const float*)d_in[5];
  const float* Wout  = (const float*)d_in[6];
  const float* bout  = (const float*)d_in[7];
  const float* ln2w  = (const float*)d_in[8];
  const float* ln2b  = (const float*)d_in[9];
  const float* W1    = (const float*)d_in[10];
  const float* b1    = (const float*)d_in[11];
  const float* W2    = (const float*)d_in[12];
  const float* b2    = (const float*)d_in[13];
  float* out = (float*)d_out;

  char* ws = (char*)d_ws;
  size_t off = 0;
  auto carve = [&](size_t bytes) -> char* {
    char* p = ws + off;
    off += (bytes + 255) & ~(size_t)255;
    return p;
  };

  _Float16* pos16  = (_Float16*)carve((size_t)B_ * N_ * D_ * 2);
  _Float16* base16 = (_Float16*)carve((size_t)N_ * D_ * 2);
  float*    sc32   = (float*)   carve((size_t)B_ * N_ * N_ * 4);
  _Float16* attn16 = (_Float16*)carve((size_t)B_ * N_ * N_ * 2);
  _Float16* xn16   = (_Float16*)carve((size_t)B_ * N_ * D_ * 2);
  _Float16* vT16   = (_Float16*)carve((size_t)B_ * INNER_ * N_ * 2);
  _Float16* o16    = (_Float16*)carve((size_t)B_ * N_ * INNER_ * 2);
  float*    r32    = (float*)   carve((size_t)B_ * N_ * D_ * 4);
  _Float16* y16    = (_Float16*)carve((size_t)B_ * N_ * D_ * 2);
  _Float16* h16    = (_Float16*)carve((size_t)B_ * N_ * MLP_ * 2);
  float*    xbuf   = (float*)   carve((size_t)B_ * N_ * D_ * 4);
  _Float16* Wv16   = (_Float16*)carve((size_t)L_ * INNER_ * D_ * 2);
  _Float16* Wo16   = (_Float16*)carve((size_t)L_ * D_ * INNER_ * 2);
  _Float16* W116   = (_Float16*)carve((size_t)L_ * MLP_ * D_ * 2);
  _Float16* W216   = (_Float16*)carve((size_t)L_ * D_ * MLP_ * 2);

  auto cvt = [&](const float* src, _Float16* dst, long n) {
    cvt_f32_f16_kernel<<<dim3(2048), dim3(256), 0, stream>>>(src, dst, n);
  };

  // f16 conversions (activations for scores + all weights, once per call)
  cvt(pos,   pos16,  (long)B_ * N_ * D_);
  cvt(basep, base16, (long)N_ * D_);
  for (int l = 0; l < L_; ++l)  // only the V slice of Wqkv is used
    cvt(Wqkv + ((long)l * 3 * INNER_ + 2 * INNER_) * D_,
        Wv16 + (long)l * INNER_ * D_, (long)INNER_ * D_);
  cvt(Wout, Wo16,  (long)L_ * D_ * INNER_);
  cvt(W1,   W116,  (long)L_ * MLP_ * D_);
  cvt(W2,   W216,  (long)L_ * D_ * MLP_);

  // Positional attention scores (shared by all layers/heads), then softmax.
  const float scale = 1.0f / sqrtf((float)D_);
  gemm_wmma_kernel<0><<<dim3(N_ / 128, N_ / 128, B_), 256, 0, stream>>>(
      pos16, base16, nullptr, nullptr, sc32, N_, N_, D_, N_, scale,
      1, (long)N_ * D_, 0, 0, 0, (long)N_ * N_, 0);
  softmax_kernel<<<B_ * N_, 256, 0, stream>>>(sc32, attn16);

  for (int l = 0; l < L_; ++l) {
    const float* xin = (l == 0) ? x : xbuf;

    // xn = LN1(x)
    ln_kernel<<<B_ * N_, 256, 0, stream>>>(xin, ln1w + l * D_, ln1b + l * D_,
                                           xn16, D_);

    // vT[b][inner][token] = (xn @ Wv^T) transposed store (per-b batch)
    gemm_wmma_kernel<2><<<dim3(INNER_ / 128, N_ / 128, B_), 256, 0, stream>>>(
        xn16, Wv16 + (long)l * INNER_ * D_, nullptr, nullptr, vT16,
        N_, INNER_, D_, /*ldc=tokens*/ N_, 1.0f,
        1, (long)N_ * D_, 0, 0, 0, (long)INNER_ * N_, 0);

    // o16[b][i][h*DH+dh] = attn[b] @ v[b,h]   (z = b*H + h)
    gemm_wmma_kernel<1><<<dim3(1, N_ / 128, B_ * H_), 256, 0, stream>>>(
        attn16, vT16, nullptr, nullptr, o16,
        N_, DH_, N_, /*ldc*/ INNER_, 1.0f,
        H_, (long)N_ * N_, 0, (long)INNER_ * N_, (long)DH_ * N_,
        (long)N_ * INNER_, (long)DH_);

    // r = o @ Wout^T + bout   (flattened M = B*N)
    gemm_wmma_kernel<3><<<dim3(D_ / 128, (B_ * N_) / 128, 1), 256, 0, stream>>>(
        o16, Wo16 + (long)l * D_ * INNER_, bout + l * D_, nullptr, r32,
        B_ * N_, D_, INNER_, D_, 1.0f, 1, 0, 0, 0, 0, 0, 0);

    // y = LN2(r)
    ln_kernel<<<B_ * N_, 256, 0, stream>>>(r32, ln2w + l * D_, ln2b + l * D_,
                                           y16, D_);

    // h = gelu(y @ W1^T + b1)
    gemm_wmma_kernel<4><<<dim3(MLP_ / 128, (B_ * N_) / 128, 1), 256, 0, stream>>>(
        y16, W116 + (long)l * MLP_ * D_, b1 + l * MLP_, nullptr, h16,
        B_ * N_, MLP_, D_, MLP_, 1.0f, 1, 0, 0, 0, 0, 0, 0);

    // x_next = h @ W2^T + b2 + r   (last layer writes d_out)
    float* dst = (l == L_ - 1) ? out : xbuf;
    gemm_wmma_kernel<5><<<dim3(D_ / 128, (B_ * N_) / 128, 1), 256, 0, stream>>>(
        h16, W216 + (long)l * D_ * MLP_, b2 + l * D_, r32, dst,
        B_ * N_, D_, MLP_, D_, 1.0f, 1, 0, 0, 0, 0, 0, 0);
  }
}
